// FasterSpiralConv_19980187861858
// MI455X (gfx1250) — matrix operand order repaired and checked by reading, not compiled, not run on hardware
//
#include <hip/hip_runtime.h>

#define B_   8
#define V_   50000
#define C_   256
#define PC_  64
#define K_   9
#define KC1_ 576           // K_ * PC_
#define TM_  64            // rows per workgroup
#define TPB_ 782           // ceil(V_/TM_)

// split-bf16 weight workspace layout (ushort offsets)
#define W1_HI_OFF 0
#define W1_LO_OFF 36864
#define W2_HI_OFF 73728
#define W2_LO_OFF 139264
#define WS_USHORTS 204800   // 409600 bytes

typedef __attribute__((ext_vector_type(16))) __bf16          v16bf;
typedef __attribute__((ext_vector_type(8)))  float           v8f;
typedef __attribute__((ext_vector_type(8)))  unsigned short  u16x8;
typedef __attribute__((ext_vector_type(4)))  float           f32x4;

// Truncation-based split: f = hi + lo + O(2^-16 * f).
// hi = f with mantissa chopped to bf16; lo = bf16-trunc(f - hi).
static __device__ __forceinline__ void split1(float f, unsigned short& h, unsigned short& l) {
  unsigned u  = __float_as_uint(f);
  unsigned uh = u & 0xFFFF0000u;
  float    r  = f - __uint_as_float(uh);
  h = (unsigned short)(u >> 16);
  l = (unsigned short)(__float_as_uint(r) >> 16);
}

// Split 8 consecutive fp32 into (hi, lo) bf16 halves, store as 16B vectors.
static __device__ __forceinline__ void split_store8(const float* __restrict__ s,
                                                    unsigned short* __restrict__ dh,
                                                    unsigned short* __restrict__ dl) {
  f32x4 f0 = *(const f32x4*)(s);
  f32x4 f1 = *(const f32x4*)(s + 4);
  u16x8 h, l;
#pragma unroll
  for (int j = 0; j < 8; ++j) {
    unsigned short hi, lo;
    split1((j < 4) ? f0[j] : f1[j - 4], hi, lo);
    h[j] = hi; l[j] = lo;
  }
  *(u16x8*)dh = h;
  *(u16x8*)dl = l;
}

// Non-temporal-load variant (streaming source, read exactly once).
static __device__ __forceinline__ void split_store8_nt(const float* __restrict__ s,
                                                       unsigned short* __restrict__ dh,
                                                       unsigned short* __restrict__ dl) {
  f32x4 f0 = __builtin_nontemporal_load((const f32x4*)(s));
  f32x4 f1 = __builtin_nontemporal_load((const f32x4*)(s + 4));
  u16x8 h, l;
#pragma unroll
  for (int j = 0; j < 8; ++j) {
    unsigned short hi, lo;
    split1((j < 4) ? f0[j] : f1[j - 4], hi, lo);
    h[j] = hi; l[j] = lo;
  }
  *(u16x8*)dh = h;
  *(u16x8*)dl = l;
}

// Copy 16 ushorts (two 16B vectors).
static __device__ __forceinline__ void copy16(const unsigned short* __restrict__ s,
                                              unsigned short* __restrict__ d) {
  u16x8 a = *(const u16x8*)(s);
  u16x8 b = *(const u16x8*)(s + 8);
  *(u16x8*)d       = a;
  *(u16x8*)(d + 8) = b;
}

// A fragment (16x32 bf16) per ISA layout:
// lane<16:  elems0-7 -> K=kbase+0..7,   elems8-15 -> K=kbase+16..23
// lane>=16: elems0-7 -> K=kbase+8..15,  elems8-15 -> K=kbase+24..31
static __device__ __forceinline__ v16bf load_frag_a(const unsigned short* __restrict__ p,
                                                    int stride, int mbase, int kbase, int lane) {
  const unsigned short* q = p + (size_t)(mbase + (lane & 15)) * stride
                              + kbase + ((lane >> 4) << 3);
  union { u16x8 u[2]; v16bf v; } r;
  r.u[0] = *(const u16x8*)(q);
  r.u[1] = *(const u16x8*)(q + 16);
  return r.v;
}

// B fragment (32x16 bf16): lane<16: N=lane, K=kbase+0..7,16..23; lane>=16: N=lane-16, K=kbase+8..15,24..31
static __device__ __forceinline__ v16bf load_frag_b(const unsigned short* __restrict__ p,
                                                    int stride, int nbase, int kbase, int lane) {
  const unsigned short* q = p + (size_t)(nbase + (lane & 15)) * stride
                              + kbase + ((lane >> 4) << 3);
  union { u16x8 u[2]; v16bf v; } r;
  r.u[0] = *(const u16x8*)(q);
  r.u[1] = *(const u16x8*)(q + 16);
  return r.v;
}

static __device__ __forceinline__ v8f wmma_bf16(v16bf a, v16bf b, v8f c) {
  return __builtin_amdgcn_wmma_f32_16x16x32_bf16(false, a, false, b, (short)0, c, false, false);
}
// split-bf16 "bf16x3": hi*hi + lo*hi + hi*lo (lo*lo negligible), fp32 accumulate
static __device__ __forceinline__ v8f mm3(v16bf ah, v16bf al, v16bf bh, v16bf bl, v8f c) {
  c = wmma_bf16(ah, bh, c);
  c = wmma_bf16(al, bh, c);
  c = wmma_bf16(ah, bl, c);
  return c;
}

// Pre-kernel: split W1/W2 into hi/lo bf16 arrays in workspace (run each launch; tiny).
__global__ __launch_bounds__(256)
void convert_w_kernel(const float* __restrict__ W1, const float* __restrict__ W2,
                      unsigned short* __restrict__ ws) {
  const int gid = blockIdx.x * 256 + threadIdx.x;   // 8 elements per thread
  if (gid < 4608) {                                  // W1: 64*576 = 36864 elems
    const int e = gid * 8;
    split_store8(W1 + e, ws + W1_HI_OFF + e, ws + W1_LO_OFF + e);
  } else if (gid < 4608 + 8192) {                    // W2: 256*256 = 65536 elems
    const int e = (gid - 4608) * 8;
    split_store8(W2 + e, ws + W2_HI_OFF + e, ws + W2_LO_OFF + e);
  }
}

// LDS map (ushorts):
//   [0,4096)      a_hi   64x64 gathered chunk      \  aliased in GEMM2 as
//   [4096,8192)   a_lo                              |  w2_hi 256x32 [0,8192)
//   [8192,12288)  w1_hi  64x64 W1 chunk             |  w2_lo 256x32 [8192,16384)
//   [12288,16384) w1_lo                            /
//   [16384,32768) cat_hi 64x256 (y1 | untouched)
//   [32768,49152) cat_lo
#define SMEM_USHORTS 49152

template <bool WP>   // WP: weights pre-split in workspace
__global__ __launch_bounds__(256)
void spiral_fused_kernel(const float* __restrict__ x,
                         const long long* __restrict__ idxs,
                         const float* __restrict__ W1,
                         const float* __restrict__ b1,
                         const float* __restrict__ W2,
                         const float* __restrict__ b2,
                         const unsigned short* __restrict__ wsp,
                         float* __restrict__ out) {
  extern __shared__ unsigned short smem[];
  unsigned short* a_hi   = smem;
  unsigned short* a_lo   = smem + 4096;
  unsigned short* w1_hi  = smem + 8192;
  unsigned short* w1_lo  = smem + 12288;
  unsigned short* w2_hi  = smem;           // aliases a_*/w1_* (phase-separated by barriers)
  unsigned short* w2_lo  = smem + 8192;
  unsigned short* cat_hi = smem + 16384;
  unsigned short* cat_lo = smem + 32768;

  const int tid  = threadIdx.x;
  const int lane = tid & 31;
  const int w    = tid >> 5;     // 8 waves
  const int mt   = w >> 1;       // M-tile 0..3 (16 rows each)
  const int wn   = w & 1;        // N half 0..1

  const int bidx = blockIdx.x / TPB_;
  const int tb   = blockIdx.x % TPB_;
  const int v0   = tb * TM_;
  const int rows_valid = min(TM_, V_ - v0);
  const int mrow = mt * 16;

  // ---- stage untouched channels 64..255 into cat (split bf16, NT loads) ----
  {
    const int r = tid >> 2, q = tid & 3;            // r: row 0..63, q: 48-ch segment
    const int vr = min(v0 + r, V_ - 1);
    const float* src = x + ((size_t)bidx * V_ + vr) * C_ + PC_ + q * 48;
    unsigned short* dh = cat_hi + r * C_ + PC_ + q * 48;
    unsigned short* dl = cat_lo + r * C_ + PC_ + q * 48;
#pragma unroll
    for (int j = 0; j < 48; j += 8) split_store8_nt(src + j, dh + j, dl + j);
  }

  // ---- GEMM1: y1[64 x 64] = gathered[64 x 576] * W1^T, chunked over 9 gather indices ----
  v8f acc0 = {}, acc1 = {};
  for (int kc = 0; kc < K_; ++kc) {
    __syncthreads();
    {
      const int r = tid >> 2, q = tid & 3;          // 16 elements each for gather + W1
      const int vr = min(v0 + r, V_ - 1);
      const long long gi = idxs[(size_t)vr * K_ + kc];
      const float* src = x + ((size_t)bidx * V_ + (size_t)gi) * C_ + q * 16;
      unsigned short* dh = a_hi + r * PC_ + q * 16;
      unsigned short* dl = a_lo + r * PC_ + q * 16;
      split_store8(src,     dh,     dl);
      split_store8(src + 8, dh + 8, dl + 8);

      if (kc + 1 < K_) {  // prefetch next chunk's gather row segment into caches
        const long long gn = idxs[(size_t)vr * K_ + kc + 1];
        __builtin_prefetch(x + ((size_t)bidx * V_ + (size_t)gn) * C_ + q * 16, 0, 1);
      }

      const int wofs = r * PC_ + q * 16;            // W1 chunk: row r, cols kc*64+q*16..
      if constexpr (WP) {
        const size_t so = (size_t)r * KC1_ + kc * PC_ + q * 16;
        copy16(wsp + W1_HI_OFF + so, w1_hi + wofs);
        copy16(wsp + W1_LO_OFF + so, w1_lo + wofs);
      } else {
        const float* wsrc = W1 + (size_t)r * KC1_ + kc * PC_ + q * 16;
        split_store8(wsrc,     w1_hi + wofs,     w1_lo + wofs);
        split_store8(wsrc + 8, w1_hi + wofs + 8, w1_lo + wofs + 8);
      }
    }
    __syncthreads();
#pragma unroll
    for (int ks = 0; ks < PC_; ks += 32) {
      v16bf ah = load_frag_a(a_hi, PC_, mrow, ks, lane);
      v16bf al = load_frag_a(a_lo, PC_, mrow, ks, lane);
      v16bf bh = load_frag_b(w1_hi, PC_, wn * 32,      ks, lane);
      v16bf bl = load_frag_b(w1_lo, PC_, wn * 32,      ks, lane);
      acc0 = mm3(ah, al, bh, bl, acc0);
      bh = load_frag_b(w1_hi, PC_, wn * 32 + 16, ks, lane);
      bl = load_frag_b(w1_lo, PC_, wn * 32 + 16, ks, lane);
      acc1 = mm3(ah, al, bh, bl, acc1);
    }
  }

  // ---- GEMM1 epilogue: + b1, write split y1 into cat[:, 0:64] ----
  {
    const int mofs = mrow + ((lane >> 4) << 3);     // D layout: VGPR r -> M=r (+8 upper lanes)
#pragma unroll
    for (int t = 0; t < 2; ++t) {
      const int n = wn * 32 + t * 16 + (lane & 15);
      const float bv = b1[n];
      v8f a = t ? acc1 : acc0;
#pragma unroll
      for (int r = 0; r < 8; ++r) {
        unsigned short hi, lo;
        split1(a[r] + bv, hi, lo);
        cat_hi[(mofs + r) * C_ + n] = hi;
        cat_lo[(mofs + r) * C_ + n] = lo;
      }
    }
  }

  // ---- GEMM2: out[64 x 256] = x_cat[64 x 256] * W2^T, W2 streamed in 32-col chunks ----
  v8f zero = {};
  v8f acc2[8];
#pragma unroll
  for (int i = 0; i < 8; ++i) acc2[i] = zero;

  for (int s = 0; s < 8; ++s) {
    __syncthreads();                                 // cat ready (s==0) / prev chunk consumed
    {
      unsigned short* wh = w2_hi + tid * 32;         // row n = tid, 32 c's
      unsigned short* wl = w2_lo + tid * 32;
      if constexpr (WP) {
        const size_t so = (size_t)tid * C_ + s * 32;
        copy16(wsp + W2_HI_OFF + so,      wh);
        copy16(wsp + W2_HI_OFF + so + 16, wh + 16);
        copy16(wsp + W2_LO_OFF + so,      wl);
        copy16(wsp + W2_LO_OFF + so + 16, wl + 16);
      } else {
        const float* wsrc = W2 + (size_t)tid * C_ + s * 32;
#pragma unroll
        for (int j = 0; j < 32; j += 8) split_store8(wsrc + j, wh + j, wl + j);
      }
    }
    __syncthreads();
    v16bf ah = load_frag_a(cat_hi, C_, mrow, s * 32, lane);
    v16bf al = load_frag_a(cat_lo, C_, mrow, s * 32, lane);
#pragma unroll
    for (int nt = 0; nt < 8; ++nt) {
      v16bf bh = load_frag_b(w2_hi, 32, wn * 128 + nt * 16, 0, lane);
      v16bf bl = load_frag_b(w2_lo, 32, wn * 128 + nt * 16, 0, lane);
      acc2[nt] = mm3(ah, al, bh, bl, acc2[nt]);
    }
  }

  // ---- GEMM2 epilogue: + b2, NT dword stores (coalesced per half-wave), tail predication ----
  {
    const int mofs = mrow + ((lane >> 4) << 3);
#pragma unroll
    for (int nt = 0; nt < 8; ++nt) {
      const int n = wn * 128 + nt * 16 + (lane & 15);
      const float bv = b2[n];
#pragma unroll
      for (int r = 0; r < 8; ++r) {
        const int m = mofs + r;
        if (m < rows_valid)
          __builtin_nontemporal_store(acc2[nt][r] + bv,
              out + ((size_t)bidx * V_ + v0 + m) * C_ + n);
      }
    }
  }
}

extern "C" void kernel_launch(void* const* d_in, const int* in_sizes, int n_in,
                              void* d_out, int out_size, void* d_ws, size_t ws_size,
                              hipStream_t stream) {
  (void)in_sizes; (void)n_in; (void)out_size;
  const float*     x   = (const float*)d_in[0];
  const long long* idx = (const long long*)d_in[1];
  const float*     W1  = (const float*)d_in[2];
  const float*     b1  = (const float*)d_in[3];
  const float*     W2  = (const float*)d_in[4];
  const float*     b2  = (const float*)d_in[5];
  float* out = (float*)d_out;

  const size_t smem_bytes = (size_t)SMEM_USHORTS * sizeof(unsigned short);  // 96 KB
  const bool wp = (d_ws != nullptr) &&
                  (ws_size >= (size_t)WS_USHORTS * sizeof(unsigned short));

  dim3 grid(B_ * TPB_);
  dim3 block(256);

  if (wp) {
    unsigned short* wsp = (unsigned short*)d_ws;
    convert_w_kernel<<<dim3(50), block, 0, stream>>>(W1, W2, wsp);
    (void)hipFuncSetAttribute(
        reinterpret_cast<const void*>(&spiral_fused_kernel<true>),
        hipFuncAttributeMaxDynamicSharedMemorySize, (int)smem_bytes);
    spiral_fused_kernel<true><<<grid, block, smem_bytes, stream>>>(
        x, idx, W1, b1, W2, b2, wsp, out);
  } else {
    (void)hipFuncSetAttribute(
        reinterpret_cast<const void*>(&spiral_fused_kernel<false>),
        hipFuncAttributeMaxDynamicSharedMemorySize, (int)smem_bytes);
    spiral_fused_kernel<false><<<grid, block, smem_bytes, stream>>>(
        x, idx, W1, b1, W2, b2, nullptr, out);
  }
}
